// GraphConstrainedAttentionLayer_58188216926419
// MI455X (gfx1250) — compile-verified, matrix operand order checked by reading, not run on hardware
//
#include <hip/hip_runtime.h>
#include <hip/hip_bf16.h>

typedef __attribute__((ext_vector_type(16))) __bf16 v16bf;
typedef __attribute__((ext_vector_type(8)))  __bf16 v8bf;
typedef __attribute__((ext_vector_type(8)))  float  v8f;

#define D_ 128
#define H_ 8
#define DH_ 16
#define FE_ 32

__device__ inline v8f wmma_bf16(v16bf a, v16bf b, v8f c) {
    return __builtin_amdgcn_wmma_f32_16x16x32_bf16(
        false, a, false, b, (short)0, c, false, false);
}

// Load a prepacked B fragment: 32 contiguous bytes per lane.
__device__ inline v16bf load_bfrag(const __bf16* __restrict__ base, int fragIdx, int lane) {
    return *(const v16bf*)(base + ((size_t)fragIdx * 32 + lane) * 16);
}

// Load an A fragment from a bf16 row-major row: halves K0..K0+7 and K0+16..K0+23.
__device__ inline v16bf load_afrag(const __bf16* __restrict__ row, int K0) {
    v8bf lo = *(const v8bf*)(row + K0);
    v8bf hi = *(const v8bf*)(row + K0 + 16);
    v16bf r;
#pragma unroll
    for (int t = 0; t < 8; ++t) { r[t] = lo[t]; r[t + 8] = hi[t]; }
    return r;
}

// ---------------------------------------------------------------------------
// Pack kernels (run once per launch; trivial traffic)
// ---------------------------------------------------------------------------
__global__ void pack_bf16(const float* __restrict__ src, __bf16* __restrict__ dst, int n) {
    int i = blockIdx.x * blockDim.x + threadIdx.x;
    if (i < n) dst[i] = (__bf16)src[i];
}

// W [K x Ncols] row-major f32 -> WMMA-B fragment-major bf16 (zero-padded cols).
// dst[((c*nk + kk)*32 + l)*16 + t] = W[kk*32 + (l>>4)*8 + (t<8?t:t+8), c*16 + (l&15)]
__global__ void pack_w_frag(const float* __restrict__ W, __bf16* __restrict__ dst,
                            int K, int Ncols, int NcolsPad) {
    int idx = blockIdx.x * blockDim.x + threadIdx.x;
    if (idx >= K * NcolsPad) return;
    const int t    = idx & 15;
    const int l    = (idx >> 4) & 31;
    const int frag = idx >> 9;
    const int nk   = K >> 5;
    const int kk   = frag % nk;
    const int c    = frag / nk;
    const int k    = kk * 32 + ((l >> 4) * 8) + (t < 8 ? t : t + 8);
    const int n    = c * 16 + (l & 15);
    dst[idx] = (__bf16)((n < Ncols) ? W[(size_t)k * Ncols + n] : 0.0f);
}

// ---------------------------------------------------------------------------
// Init: softmax max (-inf), denom (0), attention accumulator (0)
// ---------------------------------------------------------------------------
__global__ void init_buffers(float* __restrict__ mmax, float* __restrict__ denom,
                             float* __restrict__ attn, int nm, int na) {
    int i = blockIdx.x * blockDim.x + threadIdx.x;
    if (i < nm) { mmax[i] = -__builtin_inff(); denom[i] = 0.0f; }
    if (i < na) attn[i] = 0.0f;
}

// ---------------------------------------------------------------------------
// QKV projection. grid N/16, block 256 (8 waves = 8 column tiles).
// Each wave: A fragment loaded once per k-step, feeds Wq/Wk/Wv WMMAs (3x reuse).
// ---------------------------------------------------------------------------
__global__ __launch_bounds__(256) void qkv_gemm(
    const __bf16* __restrict__ hbf,
    const __bf16* __restrict__ Wqf, const __bf16* __restrict__ Wkf, const __bf16* __restrict__ Wvf,
    float* __restrict__ Qo, float* __restrict__ Ko, float* __restrict__ Vo)
{
    const int lane = threadIdx.x & 31;
    const int ctl  = threadIdx.x >> 5;          // column tile 0..7
    const int row0 = blockIdx.x * 16;
    const int m    = lane & 15;
    const int kb   = (lane >> 4) * 8;
    const __bf16* Arow = hbf + (size_t)(row0 + m) * D_;

    v8f aq = {}, ak = {}, av = {};
#pragma unroll
    for (int kk = 0; kk < 4; ++kk) {
        const v16bf a = load_afrag(Arow, kk * 32 + kb);
        const int frag = ctl * 4 + kk;
        aq = wmma_bf16(a, load_bfrag(Wqf, frag, lane), aq);
        ak = wmma_bf16(a, load_bfrag(Wkf, frag, lane), ak);
        av = wmma_bf16(a, load_bfrag(Wvf, frag, lane), av);
    }
    const int ncol = ctl * 16 + m;
#pragma unroll
    for (int r = 0; r < 8; ++r) {
        const size_t o = (size_t)(row0 + r + kb) * D_ + ncol;
        Qo[o] = aq[r]; Ko[o] = ak[r]; Vo[o] = av[r];
    }
}

// ---------------------------------------------------------------------------
// Edge bias MLP. Wave = 16 edges. L1: [16x32]@[32x128]+relu (LDS bf16 stage),
// L2: [16x128]@[128x16-padded]. Writes bias into scores[E][8].
// ---------------------------------------------------------------------------
__global__ __launch_bounds__(256) void edge_bias_mlp(
    const float* __restrict__ ef, const __bf16* __restrict__ W1f, const float* __restrict__ b1,
    const __bf16* __restrict__ W2f, const float* __restrict__ b2, float* __restrict__ scores)
{
    __shared__ __bf16 hid[8][16 * D_];
    const int lane = threadIdx.x & 31;
    const int wave = threadIdx.x >> 5;
    const int e0   = (blockIdx.x * 8 + wave) * 16;
    const int m    = lane & 15;
    const int kb   = (lane >> 4) * 8;

    // layer 1: A = edge_feat[e0+m][:]  (K = 32, single k-step)
    v16bf a;
    const float* Erow = ef + (size_t)(e0 + m) * FE_;
#pragma unroll
    for (int t = 0; t < 8; ++t) {
        a[t]     = (__bf16)Erow[kb + t];
        a[t + 8] = (__bf16)Erow[kb + 16 + t];
    }
#pragma unroll
    for (int nt = 0; nt < 8; ++nt) {
        v8f acc = {};
        acc = wmma_bf16(a, load_bfrag(W1f, nt, lane), acc);
        const int ncol = nt * 16 + m;
        const float bias1 = b1[ncol];
#pragma unroll
        for (int r = 0; r < 8; ++r) {
            float v = acc[r] + bias1;
            v = v > 0.0f ? v : 0.0f;
            hid[wave][(r + kb) * D_ + ncol] = (__bf16)v;
        }
    }
    __syncthreads();

    // layer 2: hidden @ W2 (cols 8..15 pre-zeroed in packed W2)
    v8f acc2 = {};
    const __bf16* hrow = &hid[wave][m * D_];
#pragma unroll
    for (int kk = 0; kk < 4; ++kk) {
        const v16bf a2 = load_afrag(hrow, kk * 32 + kb);
        acc2 = wmma_bf16(a2, load_bfrag(W2f, kk, lane), acc2);
    }
    const int col = m;
    if (col < H_) {
        const float bb = b2[col];
#pragma unroll
        for (int r = 0; r < 8; ++r)
            scores[(size_t)(e0 + r + kb) * H_ + col] = acc2[r] + bb;
    }
}

// ---------------------------------------------------------------------------
// Per-(edge,head) score = QK/sqrt(16) + bias, f32 atomic max via int trick.
// ---------------------------------------------------------------------------
__global__ void edge_score(const float* __restrict__ Q, const float* __restrict__ Km,
                           const int* __restrict__ esrc, const int* __restrict__ edst,
                           float* __restrict__ scores, float* __restrict__ mmax, int EH)
{
    int idx = blockIdx.x * blockDim.x + threadIdx.x;
    if (idx >= EH) return;
    const int e = idx >> 3, hd = idx & 7;
    const int s = esrc[e], d = edst[e];
    const float4* q = (const float4*)(Q  + (size_t)d * D_ + hd * DH_);
    const float4* k = (const float4*)(Km + (size_t)s * D_ + hd * DH_);
    float dot = 0.0f;
#pragma unroll
    for (int j = 0; j < 4; ++j) {
        float4 qv = q[j], kv = k[j];
        dot += qv.x * kv.x + qv.y * kv.y + qv.z * kv.z + qv.w * kv.w;
    }
    const float sc = dot * 0.25f + scores[idx];
    scores[idx] = sc;
    float* addr = mmax + (size_t)d * H_ + hd;
    if (sc >= 0.0f) atomicMax((int*)addr, __float_as_int(sc));
    else            atomicMin((unsigned int*)addr, (unsigned int)__float_as_int(sc));
}

__global__ void edge_exp(const int* __restrict__ edst, const float* __restrict__ mmax,
                         float* __restrict__ scores, float* __restrict__ denom, int EH)
{
    int idx = blockIdx.x * blockDim.x + threadIdx.x;
    if (idx >= EH) return;
    const int e = idx >> 3, hd = idx & 7;
    const int d = edst[e];
    const float ex = __expf(scores[idx] - mmax[(size_t)d * H_ + hd]);
    scores[idx] = ex;
    atomicAdd(denom + (size_t)d * H_ + hd, ex);
}

__global__ void edge_scatter(const int* __restrict__ esrc, const int* __restrict__ edst,
                             const float* __restrict__ scores, const float* __restrict__ denom,
                             const float* __restrict__ V, float* __restrict__ attn, int EH)
{
    int idx = blockIdx.x * blockDim.x + threadIdx.x;
    if (idx >= EH) return;
    const int e = idx >> 3, hd = idx & 7;
    const int s = esrc[e], d = edst[e];
    const float alpha = scores[idx] / denom[(size_t)d * H_ + hd];
    const float* v = V    + (size_t)s * D_ + hd * DH_;
    float*       o = attn + (size_t)d * D_ + hd * DH_;
#pragma unroll
    for (int j = 0; j < DH_; ++j) atomicAdd(o + j, alpha * v[j]);
}

// ---------------------------------------------------------------------------
// out = prelu(attn @ Wo + bo + h). grid N/16, block 256 (8 waves = 8 col tiles).
// A is the f32 atomic accumulator -> convert in-register; B prepacked bf16.
// ---------------------------------------------------------------------------
__global__ __launch_bounds__(256) void out_gemm(
    const float* __restrict__ attn, const __bf16* __restrict__ Wof,
    const float* __restrict__ bo, const float* __restrict__ hin,
    const float* __restrict__ prelu_a, float* __restrict__ out)
{
    const int lane = threadIdx.x & 31;
    const int ctl  = threadIdx.x >> 5;
    const int row0 = blockIdx.x * 16;
    const int m    = lane & 15;
    const int kb   = (lane >> 4) * 8;
    const float* Arow = attn + (size_t)(row0 + m) * D_;

    v8f acc = {};
#pragma unroll
    for (int kk = 0; kk < 4; ++kk) {
        const int K0 = kk * 32 + kb;
        v16bf a;
#pragma unroll
        for (int t = 0; t < 8; ++t) {
            a[t]     = (__bf16)Arow[K0 + t];
            a[t + 8] = (__bf16)Arow[K0 + 16 + t];
        }
        acc = wmma_bf16(a, load_bfrag(Wof, ctl * 4 + kk, lane), acc);
    }
    const int ncol = ctl * 16 + m;
    const float ap = prelu_a[0];
    const float bb = bo[ncol];
#pragma unroll
    for (int r = 0; r < 8; ++r) {
        const int row = row0 + r + kb;
        const float pre = acc[r] + bb + hin[(size_t)row * D_ + ncol];
        out[(size_t)row * D_ + ncol] = (pre >= 0.0f) ? pre : ap * pre;
    }
}

// ---------------------------------------------------------------------------
extern "C" void kernel_launch(void* const* d_in, const int* in_sizes, int n_in,
                              void* d_out, int out_size, void* d_ws, size_t ws_size,
                              hipStream_t stream) {
    const float* hin  = (const float*)d_in[0];
    const float* ef   = (const float*)d_in[1];
    const int*   esrc = (const int*)d_in[2];
    const int*   edst = (const int*)d_in[3];
    const float* Wq   = (const float*)d_in[4];
    const float* Wk   = (const float*)d_in[5];
    const float* Wv   = (const float*)d_in[6];
    const float* W1   = (const float*)d_in[7];
    const float* b1   = (const float*)d_in[8];
    const float* W2   = (const float*)d_in[9];
    const float* b2   = (const float*)d_in[10];
    const float* Wo   = (const float*)d_in[11];
    const float* bo   = (const float*)d_in[12];
    const float* pa   = (const float*)d_in[13];
    float* out = (float*)d_out;

    const int N  = in_sizes[0] / D_;   // 50000
    const int E  = in_sizes[2];        // 800000
    const int EH = E * H_;
    const size_t ND = (size_t)N * D_;

    // f32 workspace region
    float* ws     = (float*)d_ws;
    float* Qp     = ws;
    float* Kp     = Qp + ND;
    float* Vp     = Kp + ND;
    float* attn   = Vp + ND;
    float* scores = attn + ND;                 // E*H
    float* mmax   = scores + (size_t)E * H_;   // N*H
    float* denom  = mmax + (size_t)N * H_;     // N*H
    // bf16 workspace region (32B-aligned by construction)
    __bf16* hbf = (__bf16*)(denom + (size_t)N * H_);  // N*D
    __bf16* Wqf = hbf + ND;                    // 128*128
    __bf16* Wkf = Wqf + D_ * D_;
    __bf16* Wvf = Wkf + D_ * D_;
    __bf16* Wof = Wvf + D_ * D_;
    __bf16* W1f = Wof + D_ * D_;               // 32*128
    __bf16* W2f = W1f + FE_ * D_;              // 128*16 (padded)

    // packing passes
    pack_bf16<<<dim3(((int)ND + 255) / 256), 256, 0, stream>>>(hin, hbf, (int)ND);
    pack_w_frag<<<dim3(64), 256, 0, stream>>>(Wq, Wqf, D_, D_, D_);
    pack_w_frag<<<dim3(64), 256, 0, stream>>>(Wk, Wkf, D_, D_, D_);
    pack_w_frag<<<dim3(64), 256, 0, stream>>>(Wv, Wvf, D_, D_, D_);
    pack_w_frag<<<dim3(64), 256, 0, stream>>>(Wo, Wof, D_, D_, D_);
    pack_w_frag<<<dim3(16), 256, 0, stream>>>(W1, W1f, FE_, D_, D_);
    pack_w_frag<<<dim3(8),  256, 0, stream>>>(W2, W2f, D_, H_, 16);

    init_buffers<<<dim3((N * D_ + 255) / 256), 256, 0, stream>>>(mmax, denom, attn, N * H_, N * D_);
    qkv_gemm<<<dim3(N / 16), 256, 0, stream>>>(hbf, Wqf, Wkf, Wvf, Qp, Kp, Vp);
    edge_bias_mlp<<<dim3(E / 128), 256, 0, stream>>>(ef, W1f, b1, W2f, b2, scores);
    edge_score<<<dim3((EH + 255) / 256), 256, 0, stream>>>(Qp, Kp, esrc, edst, scores, mmax, EH);
    edge_exp<<<dim3((EH + 255) / 256), 256, 0, stream>>>(edst, mmax, scores, denom, EH);
    edge_scatter<<<dim3((EH + 255) / 256), 256, 0, stream>>>(esrc, edst, scores, denom, Vp, attn, EH);
    out_gemm<<<dim3(N / 16), 256, 0, stream>>>(attn, Wof, bo, hin, pa, out);
}
